// MultiHeadSelfAttention_62079457296413
// MI455X (gfx1250) — matrix operand order repaired and checked by reading, not compiled, not run on hardware
//
#include <hip/hip_runtime.h>
#include <hip/hip_bf16.h>

// Problem constants
#define BB 2
#define SS 2048
#define DD 1024
#define HH 16
#define DHD 64
#define MM (BB * SS)   // 4096 rows

typedef __attribute__((ext_vector_type(16))) __bf16 v16bf;
typedef __attribute__((ext_vector_type(8)))  __bf16 v8bf;
typedef __attribute__((ext_vector_type(4)))  __bf16 v4bf;
typedef __attribute__((ext_vector_type(8)))  float  v8f;
typedef __attribute__((ext_vector_type(4)))  unsigned int v4u;
typedef __attribute__((ext_vector_type(8)))  int v8i;
typedef __attribute__((ext_vector_type(4)))  int v4i;

#define WMMA_BF16(a, b, c) \
  __builtin_amdgcn_wmma_f32_16x16x32_bf16(false, (a), false, (b), (short)0, (c), false, false)

#if __has_builtin(__builtin_amdgcn_tensor_load_to_lds)
#define USE_TDM 1
#else
#define USE_TDM 0
#endif

// Wave-relative LDS byte offset of a pointer to __shared__ (flat LDS address
// is {SHARED_BASE, offset[31:0]}, ISA 10.2 — low 32 bits are the DS address).
__device__ __forceinline__ unsigned lds_off_of(const void* p) {
  return (unsigned)(uintptr_t)p;
}

// ---- Fragment loaders (ISA 7.12.2 layouts, wave32) -------------------------
// A (16x32 bf16): lane l holds row (l&15); elems 0..7 -> k=(l>>4)*8+j,
//                 elems 8..15 -> k=16+(l>>4)*8+(j-8). Two 16B loads.
__device__ __forceinline__ v16bf load_a_frag(const __bf16* base, int ld, int row0, int k0) {
  const int lane = threadIdx.x & 31;
  const __bf16* p = base + (size_t)(row0 + (lane & 15)) * ld + k0 + ((lane >> 4) << 3);
  v8bf lo = *(const v8bf*)(p);
  v8bf hi = *(const v8bf*)(p + 16);
  return __builtin_shufflevector(lo, hi, 0,1,2,3,4,5,6,7,8,9,10,11,12,13,14,15);
}

// B (32x16 bf16): lane l holds col (l&15); elems 0..7 -> k=(l>>4)*8+j,
//                 elems 8..15 -> k=16+(l>>4)*8+(j-8). Two 16B loads.
__device__ __forceinline__ v16bf load_b_frag(const __bf16* base, int ld, int col0, int k0) {
  const int lane = threadIdx.x & 31;
  const __bf16* p = base + (size_t)(col0 + (lane & 15)) * ld + k0 + ((lane >> 4) << 3);
  v8bf lo = *(const v8bf*)(p);
  v8bf hi = *(const v8bf*)(p + 16);
  return __builtin_shufflevector(lo, hi, 0,1,2,3,4,5,6,7,8,9,10,11,12,13,14,15);
}

// LDS tile fragment loads via inline asm ds_load_b128. The tiles are written
// by the Tensor Data Mover, invisible to the compiler, so the reads must be
// opaque (asm) and must end with s_wait_dscnt before the data is consumed.
// Tiles are [row][32] bf16 row-major (64 B per row).
__device__ __forceinline__ v16bf lds_a_frag(const __bf16* base) {
  const int lane = threadIdx.x & 31;
  const unsigned off = lds_off_of(base + (lane & 15) * 32 + ((lane >> 4) << 3));
  v8bf lo, hi;
  asm volatile("ds_load_b128 %0, %2\n\t"
               "ds_load_b128 %1, %2 offset:32\n\t"
               "s_wait_dscnt 0"
               : "=&v"(lo), "=&v"(hi)
               : "v"(off));
  return __builtin_shufflevector(lo, hi, 0,1,2,3,4,5,6,7,8,9,10,11,12,13,14,15);
}
__device__ __forceinline__ v16bf lds_b_frag(const __bf16* base) {
  const int lane = threadIdx.x & 31;
  const unsigned off = lds_off_of(base + (lane & 15) * 32 + ((lane >> 4) << 3));
  v8bf lo, hi;
  asm volatile("ds_load_b128 %0, %2\n\t"
               "ds_load_b128 %1, %2 offset:32\n\t"
               "s_wait_dscnt 0"
               : "=&v"(lo), "=&v"(hi)
               : "v"(off));
  return __builtin_shufflevector(lo, hi, 0,1,2,3,4,5,6,7,8,9,10,11,12,13,14,15);
}

// ---- Tensor Data Mover: 2-D tile (rows x 32 bf16) global -> LDS ------------
// D# per CDNA5 ISA: group0 = {count|flags, lds_addr,
// global_addr lo, global_addr hi | type=2}; group1 packs data_size=2B,
// tensor dims (set huge: tiles are always in-bounds), tile dims, row stride.
#if USE_TDM
__device__ __forceinline__ void tdm_load_tile(unsigned lds_off, const __bf16* gptr,
                                              unsigned tile_rows, unsigned row_stride) {
  unsigned long long ga = (unsigned long long)(uintptr_t)gptr;
  v4u g0;
  g0[0] = 1u;                                         // count=1, user mode
  g0[1] = lds_off;                                    // lds_addr (bytes)
  g0[2] = (unsigned)(ga & 0xffffffffu);               // global_addr[31:0]
  g0[3] = (unsigned)((ga >> 32) & 0x01ffffffu) | 0x80000000u;  // addr[56:32] | type=2
  const unsigned d0 = 1u << 30, d1 = 1u << 30;        // tensor dims: no clipping
  v8i g1;
  g1[0] = (int)(1u << 16);                            // data_size=1 -> 2 bytes
  g1[1] = (int)((d0 & 0xffffu) << 16);                // tensor_dim0[15:0]
  g1[2] = (int)((d0 >> 16) | ((d1 & 0xffffu) << 16)); // dim0[31:16] | dim1[15:0]
  g1[3] = (int)((d1 >> 16) | (32u << 16));            // dim1[31:16] | tile_dim0=32
  g1[4] = (int)(tile_rows & 0xffffu);                 // tile_dim1 (tile_dim2=0)
  g1[5] = (int)row_stride;                            // tensor_dim0_stride[31:0]
  g1[6] = 0;                                          // stride hi | dim1_stride lo
  g1[7] = 0;
  v4i z4 = {0, 0, 0, 0};
  v8i z8 = {0, 0, 0, 0, 0, 0, 0, 0};
  __builtin_amdgcn_tensor_load_to_lds(g0, g1, z4, z4, z8, 0);
}
#endif

// ---- Converters ------------------------------------------------------------
__global__ __launch_bounds__(256) void cvt_x_kernel(const float* __restrict__ x,
                                                    __bf16* __restrict__ xb) {
  int i = (blockIdx.x * 256 + threadIdx.x) * 4;
  float4 v = *(const float4*)(x + i);
  v4bf o = {(__bf16)v.x, (__bf16)v.y, (__bf16)v.z, (__bf16)v.w};
  *(v4bf*)(xb + i) = o;
}

// Transpose-convert: Wt[n][k] = (bf16)W[k][n]  (coalesced writes)
__global__ __launch_bounds__(256) void cvt_w_kernel(
    const float* __restrict__ Wq, const float* __restrict__ Wk,
    const float* __restrict__ Wv, const float* __restrict__ Wo,
    __bf16* __restrict__ Wqt, __bf16* __restrict__ Wkt,
    __bf16* __restrict__ Wvt, __bf16* __restrict__ Wot) {
  int idx = blockIdx.x * 256 + threadIdx.x;
  int z = blockIdx.y;
  const float* W = (z == 0) ? Wq : (z == 1) ? Wk : (z == 2) ? Wv : Wo;
  __bf16* Wt = (z == 0) ? Wqt : (z == 1) ? Wkt : (z == 2) ? Wvt : Wot;
  int n = idx >> 10, k = idx & (DD - 1);
  Wt[(size_t)n * DD + k] = (__bf16)W[(size_t)k * DD + n];
}

// ---- Shared GEMM core: 128x128 block tile, TDM double-buffered LDS ---------
// 8 waves: mw = wv&3 (32 rows each), nw = wv>>2 (64 cols each).
struct SmemTiles {
  __bf16 A[2][128][32];
  __bf16 B[2][128][32];
};

__device__ __forceinline__ void gemm_core(SmemTiles& sm,
                                          const __bf16* __restrict__ Ag,
                                          const __bf16* __restrict__ Bg,
                                          int m0, int n0, v8f acc[2][4]) {
  const int mw = (threadIdx.x >> 5) & 3;
  const int nw = (threadIdx.x >> 7);
  const bool issuer = (threadIdx.x < 32);

#if USE_TDM
  if (issuer) {
    tdm_load_tile(lds_off_of(&sm.A[0][0][0]), Ag + (size_t)m0 * DD, 128, DD);
    tdm_load_tile(lds_off_of(&sm.B[0][0][0]), Bg + (size_t)n0 * DD, 128, DD);
  }
  int buf = 0;
  for (int kk = 0; kk < DD; kk += 32, buf ^= 1) {
    if (issuer) {
      if (kk + 32 < DD) {
        tdm_load_tile(lds_off_of(&sm.A[buf ^ 1][0][0]),
                      Ag + (size_t)m0 * DD + kk + 32, 128, DD);
        tdm_load_tile(lds_off_of(&sm.B[buf ^ 1][0][0]),
                      Bg + (size_t)n0 * DD + kk + 32, 128, DD);
        __builtin_amdgcn_s_wait_tensorcnt((short)2);  // current buffer complete
      } else {
        __builtin_amdgcn_s_wait_tensorcnt((short)0);
      }
    }
    __syncthreads();
    v16bf a0 = lds_a_frag(&sm.A[buf][mw * 32][0]);
    v16bf a1 = lds_a_frag(&sm.A[buf][mw * 32 + 16][0]);
#pragma unroll
    for (int f = 0; f < 4; ++f) {
      v16bf b = lds_b_frag(&sm.B[buf][nw * 64 + f * 16][0]);
      acc[0][f] = WMMA_BF16(a0, b, acc[0][f]);
      acc[1][f] = WMMA_BF16(a1, b, acc[1][f]);
    }
    __syncthreads();
  }
#else
  for (int kk = 0; kk < DD; kk += 32) {
    // cooperative staging fallback: 256 threads x 16B chunks
    for (int i = threadIdx.x; i < 512; i += 256) {
      int row = i >> 2, c = (i & 3) * 8;
      *(v8bf*)(&sm.A[0][row][c]) = *(const v8bf*)(Ag + (size_t)(m0 + row) * DD + kk + c);
      *(v8bf*)(&sm.B[0][row][c]) = *(const v8bf*)(Bg + (size_t)(n0 + row) * DD + kk + c);
    }
    __syncthreads();
    v16bf a0 = lds_a_frag(&sm.A[0][mw * 32][0]);
    v16bf a1 = lds_a_frag(&sm.A[0][mw * 32 + 16][0]);
#pragma unroll
    for (int f = 0; f < 4; ++f) {
      v16bf b = lds_b_frag(&sm.B[0][nw * 64 + f * 16][0]);
      acc[0][f] = WMMA_BF16(a0, b, acc[0][f]);
      acc[1][f] = WMMA_BF16(a1, b, acc[1][f]);
    }
    __syncthreads();
  }
#endif
}

// ---- QKV projection: z==0 Q [B,H,S,64], z==1 K [B,H,S,64], z==2 V^T [B,H,64,S]
__global__ __launch_bounds__(256) void qkv_gemm_kernel(
    const __bf16* __restrict__ xb,
    const __bf16* __restrict__ Wqt, const __bf16* __restrict__ Wkt,
    const __bf16* __restrict__ Wvt,
    const float* __restrict__ bq, const float* __restrict__ bk,
    const float* __restrict__ bv,
    __bf16* __restrict__ Q, __bf16* __restrict__ K, __bf16* __restrict__ Vt) {
  __shared__ SmemTiles sm;
  const int lane = threadIdx.x & 31;
  const int mw = (threadIdx.x >> 5) & 3;
  const int nw = (threadIdx.x >> 7);
  const int m0 = blockIdx.y * 128;
  const int n0 = blockIdx.x * 128;
  const int z = blockIdx.z;
  const __bf16* W = (z == 0) ? Wqt : (z == 1) ? Wkt : Wvt;
  const float* bias = (z == 0) ? bq : (z == 1) ? bk : bv;

  v8f acc[2][4] = {};
  gemm_core(sm, xb, W, m0, n0, acc);

  const int rb = (lane >> 4) * 8;
  const int cl = lane & 15;
#pragma unroll
  for (int t = 0; t < 2; ++t) {
#pragma unroll
    for (int f = 0; f < 4; ++f) {
      const int col = n0 + nw * 64 + f * 16 + cl;
      const int h = col >> 6, dh = col & (DHD - 1);
      const float bcol = bias[col];
#pragma unroll
      for (int r = 0; r < 8; ++r) {
        const int m = m0 + mw * 32 + t * 16 + rb + r;
        const int bidx = m >> 11, s = m & (SS - 1);
        const float v = acc[t][f][r] + bcol;
        if (z == 0)
          Q[((((size_t)bidx * HH + h) * SS + s) << 6) + dh] = (__bf16)v;
        else if (z == 1)
          K[((((size_t)bidx * HH + h) * SS + s) << 6) + dh] = (__bf16)v;
        else
          Vt[(((size_t)bidx * HH + h) * DHD + dh) * SS + s] = (__bf16)v;
      }
    }
  }
}

// ---- Flash attention: one wave = one 16-query tile of one head -------------
__global__ __launch_bounds__(256) void flash_attn_kernel(
    const __bf16* __restrict__ Q, const __bf16* __restrict__ K,
    const __bf16* __restrict__ Vt, __bf16* __restrict__ attn) {
  __shared__ __align__(16) __bf16 Plds[8][16][32];
  const int lane = threadIdx.x & 31;
  const int wv = threadIdx.x >> 5;
  const int gw = blockIdx.x * 8 + wv;            // 0 .. 4095
  const int qt = gw & (SS / 16 - 1);
  const int h  = (gw >> 7) & (HH - 1);
  const int b  = gw >> 11;
  const __bf16* Qh = Q + (((size_t)b * HH + h) * SS) * DHD;
  const __bf16* Kh = K + (((size_t)b * HH + h) * SS) * DHD;
  const __bf16* Vh = Vt + (((size_t)b * HH + h) * DHD) * SS;
  const int q0 = qt * 16;

  const v16bf aq0 = load_a_frag(Qh, DHD, q0, 0);
  const v16bf aq1 = load_a_frag(Qh, DHD, q0, 32);

  v8f o0 = {}, o1 = {}, o2 = {}, o3 = {};
  float mrow[8], lrow[8];
#pragma unroll
  for (int r = 0; r < 8; ++r) { mrow[r] = -1e30f; lrow[r] = 0.0f; }
  const float scale = 0.125f;                    // 1/sqrt(64)
  const int rb = (lane >> 4) * 8;
  const int cl = lane & 15;

  for (int kc = 0; kc < SS; kc += 32) {
    v16bf bk00 = load_b_frag(Kh, DHD, kc,      0);
    v16bf bk01 = load_b_frag(Kh, DHD, kc,      32);
    v16bf bk10 = load_b_frag(Kh, DHD, kc + 16, 0);
    v16bf bk11 = load_b_frag(Kh, DHD, kc + 16, 32);
    v8f s0 = {}, s1 = {};
    s0 = WMMA_BF16(aq0, bk00, s0);
    s0 = WMMA_BF16(aq1, bk01, s0);
    s1 = WMMA_BF16(aq0, bk10, s1);
    s1 = WMMA_BF16(aq1, bk11, s1);

    float p0[8], p1[8];
#pragma unroll
    for (int r = 0; r < 8; ++r) {
      float a0 = s0[r] * scale, a1 = s1[r] * scale;
      float cm = fmaxf(a0, a1);
#pragma unroll
      for (int msk = 1; msk < 16; msk <<= 1) cm = fmaxf(cm, __shfl_xor(cm, msk, 16));
      const float mnew  = fmaxf(mrow[r], cm);
      const float alpha = __expf(mrow[r] - mnew);
      const float e0 = __expf(a0 - mnew);
      const float e1 = __expf(a1 - mnew);
      float rs = e0 + e1;
#pragma unroll
      for (int msk = 1; msk < 16; msk <<= 1) rs += __shfl_xor(rs, msk, 16);
      lrow[r] = lrow[r] * alpha + rs;
      mrow[r] = mnew;
      o0[r] *= alpha; o1[r] *= alpha; o2[r] *= alpha; o3[r] *= alpha;
      p0[r] = e0; p1[r] = e1;
    }

#pragma unroll
    for (int r = 0; r < 8; ++r) {
      Plds[wv][rb + r][cl]      = (__bf16)p0[r];
      Plds[wv][rb + r][16 + cl] = (__bf16)p1[r];
    }
    asm volatile("s_wait_dscnt 0" ::: "memory");
    v8bf lo = *(const v8bf*)&Plds[wv][lane & 15][(lane >> 4) << 3];
    v8bf hi = *(const v8bf*)&Plds[wv][lane & 15][16 + ((lane >> 4) << 3)];
    v16bf ap = __builtin_shufflevector(lo, hi, 0,1,2,3,4,5,6,7,8,9,10,11,12,13,14,15);

    v16bf bv0 = load_b_frag(Vh, SS, 0,  kc);
    v16bf bv1 = load_b_frag(Vh, SS, 16, kc);
    v16bf bv2 = load_b_frag(Vh, SS, 32, kc);
    v16bf bv3 = load_b_frag(Vh, SS, 48, kc);
    o0 = WMMA_BF16(ap, bv0, o0);
    o1 = WMMA_BF16(ap, bv1, o1);
    o2 = WMMA_BF16(ap, bv2, o2);
    o3 = WMMA_BF16(ap, bv3, o3);
  }

#pragma unroll
  for (int r = 0; r < 8; ++r) {
    const float inv = 1.0f / lrow[r];
    const size_t row = (size_t)b * SS + q0 + rb + r;
    __bf16* dst = attn + row * DD + h * DHD + cl;
    dst[0]  = (__bf16)(o0[r] * inv);
    dst[16] = (__bf16)(o1[r] * inv);
    dst[32] = (__bf16)(o2[r] * inv);
    dst[48] = (__bf16)(o3[r] * inv);
  }
}

// ---- Output projection GEMM (fp32 out + bias) ------------------------------
__global__ __launch_bounds__(256) void out_gemm_kernel(
    const __bf16* __restrict__ attn, const __bf16* __restrict__ Wot,
    const float* __restrict__ bo, float* __restrict__ out) {
  __shared__ SmemTiles sm;
  const int lane = threadIdx.x & 31;
  const int mw = (threadIdx.x >> 5) & 3;
  const int nw = (threadIdx.x >> 7);
  const int m0 = blockIdx.y * 128;
  const int n0 = blockIdx.x * 128;

  v8f acc[2][4] = {};
  gemm_core(sm, attn, Wot, m0, n0, acc);

  const int rb = (lane >> 4) * 8;
  const int cl = lane & 15;
#pragma unroll
  for (int t = 0; t < 2; ++t) {
#pragma unroll
    for (int f = 0; f < 4; ++f) {
      const int col = n0 + nw * 64 + f * 16 + cl;
      const float bcol = bo[col];
#pragma unroll
      for (int r = 0; r < 8; ++r) {
        const size_t row = m0 + mw * 32 + t * 16 + rb + r;
        out[row * DD + col] = acc[t][f][r] + bcol;
      }
    }
  }
}

// ---- Host launcher ---------------------------------------------------------
extern "C" void kernel_launch(void* const* d_in, const int* in_sizes, int n_in,
                              void* d_out, int out_size, void* d_ws, size_t ws_size,
                              hipStream_t stream) {
  const float* x  = (const float*)d_in[0];
  const float* Wq = (const float*)d_in[1];
  const float* bq = (const float*)d_in[2];
  const float* Wk = (const float*)d_in[3];
  const float* bk = (const float*)d_in[4];
  const float* Wv = (const float*)d_in[5];
  const float* bv = (const float*)d_in[6];
  const float* Wo = (const float*)d_in[7];
  const float* bo = (const float*)d_in[8];
  float* out = (float*)d_out;

  char* ws = (char*)d_ws;
  size_t off = 0;
  auto alloc = [&](size_t bytes) {
    char* p = ws + off;
    off += (bytes + 255) & ~(size_t)255;
    return p;
  };
  __bf16* xb   = (__bf16*)alloc((size_t)MM * DD * 2);   // 8 MB
  __bf16* Wqt  = (__bf16*)alloc((size_t)DD * DD * 2);   // 2 MB
  __bf16* Wkt  = (__bf16*)alloc((size_t)DD * DD * 2);
  __bf16* Wvt  = (__bf16*)alloc((size_t)DD * DD * 2);
  __bf16* Wot  = (__bf16*)alloc((size_t)DD * DD * 2);
  __bf16* Qb   = (__bf16*)alloc((size_t)MM * DD * 2);   // 8 MB [B,H,S,64]
  __bf16* Kb   = (__bf16*)alloc((size_t)MM * DD * 2);   // 8 MB [B,H,S,64]
  __bf16* Vtb  = (__bf16*)alloc((size_t)MM * DD * 2);   // 8 MB [B,H,64,S]
  __bf16* attn = (__bf16*)alloc((size_t)MM * DD * 2);   // 8 MB [B,S,D]

  cvt_x_kernel<<<(MM * DD) / (256 * 4), 256, 0, stream>>>(x, xb);
  cvt_w_kernel<<<dim3((DD * DD) / 256, 4), 256, 0, stream>>>(
      Wq, Wk, Wv, Wo, Wqt, Wkt, Wvt, Wot);
  qkv_gemm_kernel<<<dim3(DD / 128, MM / 128, 3), 256, 0, stream>>>(
      xb, Wqt, Wkt, Wvt, bq, bk, bv, Qb, Kb, Vtb);
  flash_attn_kernel<<<(BB * HH * (SS / 16)) / 8, 256, 0, stream>>>(Qb, Kb, Vtb, attn);
  out_gemm_kernel<<<dim3(DD / 128, MM / 128), 256, 0, stream>>>(attn, Wot, bo, out);
}
